// Glm4Attention_30305289241161
// MI455X (gfx1250) — matrix operand order repaired and checked by reading, not compiled, not run on hardware
//
#include <hip/hip_runtime.h>
#include <hip/hip_bf16.h>

// ---------------- types / constants ----------------
typedef __bf16          v16bf __attribute__((ext_vector_type(16)));
typedef unsigned short  v16u  __attribute__((ext_vector_type(16)));
typedef float           v8f   __attribute__((ext_vector_type(8)));

static constexpr int L_   = 1024;
static constexpr int P_   = 1024;
static constexpr int HID_ = 4096;
static constexpr int H_   = 32;
static constexpr int KVH_ = 2;
static constexpr int D_   = 128;
static constexpr int ROT_ = 64;
static constexpr int KTOT_ = P_ + L_;      // 2048
static constexpr int REP_  = H_ / KVH_;    // 16
#define SCALE_F 0.08838834764831845f       // 128^-0.5
#define NEG_BIG (-1.0e9f)

#define DEV __device__ __forceinline__

DEV unsigned short f2bf(float f) {
  union { float f; unsigned u; } x; x.f = f;
  unsigned r = x.u + 0x7FFFu + ((x.u >> 16) & 1u);   // round-to-nearest-even
  return (unsigned short)(r >> 16);
}
DEV float bf2f(unsigned short u) {
  union { unsigned u; float f; } x; x.u = ((unsigned)u) << 16;
  return x.f;
}

DEV v16u make_v16u(uint4 lo, uint4 hi) {
  union { uint4 q[2]; v16u v; } u;
  u.q[0] = lo; u.q[1] = hi;
  return u.v;
}

// A-operand 16-half chunk for one K=32 step: per-lane bytes are two contiguous
// 16B groups at +0 and +16 (voff already includes row*stride + 32*half).
DEV v16u load_a16(const char* base, unsigned voff, int imm) {
  const uint4 lo = *(const uint4*)(base + voff + imm);
  const uint4 hi = *(const uint4*)(base + voff + imm + 16);
  return make_v16u(lo, hi);
}
// B-operand: 32 contiguous bytes (16 K-elements) per lane.
DEV v16u load_b16(const char* base, unsigned voff, int imm) {
  const uint4 lo = *(const uint4*)(base + voff + imm);
  const uint4 hi = *(const uint4*)(base + voff + imm + 16);
  return make_v16u(lo, hi);
}

DEV v8f wmma_bf16(v16u a, v16u b, v8f c) {
  return __builtin_amdgcn_wmma_f32_16x16x32_bf16(
      false, __builtin_bit_cast(v16bf, a),
      false, __builtin_bit_cast(v16bf, b),
      (short)0, c, false, false);
}

// ---------------- staging: f32 -> bf16 ----------------
__global__ void cvt_bf16_kernel(const float* __restrict__ src,
                                unsigned short* __restrict__ dst, int n) {
  for (int i = blockIdx.x * blockDim.x + threadIdx.x; i < n;
       i += gridDim.x * blockDim.x)
    dst[i] = f2bf(src[i]);
}

// W [K][N] f32 row-major -> Wt [N][K] bf16 (N-major for WMMA B). N = 1<<nsh.
__global__ void cvt_bf16_T_kernel(const float* __restrict__ src,
                                  unsigned short* __restrict__ dst,
                                  int K, int nsh) {
  const int total = K << nsh;
  const int nmask = (1 << nsh) - 1;
  for (int i = blockIdx.x * blockDim.x + threadIdx.x; i < total;
       i += gridDim.x * blockDim.x) {
    const int k = i >> nsh, n = i & nmask;
    dst[((size_t)n << 0) * K + k] = f2bf(src[i]);
  }
}

// ---------------- generic bf16 WMMA GEMM ----------------
// C[M,N] = A[M,K] @ Bt[N,K]^T + bias ; one wave computes a 32x128 tile.
// All hot loads: uniform scalar base (advanced by 64B/k-step) + invariant
// per-lane voffset + immediate -> pure saddr global_load_b128, no VALU addr.
__global__ __launch_bounds__(32)
void gemm_bf16_kernel(const unsigned short* __restrict__ A,
                      const unsigned short* __restrict__ Bt,
                      const float* __restrict__ bias,
                      unsigned short* __restrict__ Cb,
                      float* __restrict__ Cf,
                      int M, int N, int K) {
  const int lane = threadIdx.x;
  const int half = lane >> 4, mr = lane & 15;
  const int n0 = blockIdx.x * 128;
  const int m0 = blockIdx.y * 32;

  const unsigned a0 = (unsigned)((m0 + mr) * K) * 2u + 32u * (unsigned)half;
  const unsigned a1 = a0 + (unsigned)(16 * K) * 2u;
  unsigned bo[8];
#pragma unroll
  for (int t = 0; t < 8; ++t)
    bo[t] = (unsigned)((n0 + t * 16 + mr) * K) * 2u + 32u * (unsigned)half;

  const char* Ab = (const char*)A;
  const char* Bb = (const char*)Bt;

  v8f acc[2][8] = {};
  for (int k0 = 0; k0 < K; k0 += 32) {
    const char* up = Ab + (size_t)k0 * 2;   // uniform, +64B per step
    const char* bp = Bb + (size_t)k0 * 2;
    const v16u av0 = load_a16(up, a0, 0);
    const v16u av1 = load_a16(up, a1, 0);
#pragma unroll
    for (int t = 0; t < 8; ++t) {
      const v16u bv = load_b16(bp, bo[t], 0);
      acc[0][t] = wmma_bf16(av0, bv, acc[0][t]);
      acc[1][t] = wmma_bf16(av1, bv, acc[1][t]);
    }
  }
#pragma unroll
  for (int t = 0; t < 8; ++t) {
    const int n = n0 + t * 16 + mr;
    const float bb = bias ? bias[n] : 0.0f;
#pragma unroll
    for (int ms = 0; ms < 2; ++ms)
#pragma unroll
      for (int r = 0; r < 8; ++r) {
        const int grow = m0 + ms * 16 + r + 8 * half;
        const float val = acc[ms][t][r] + bb;
        if (Cf) Cf[(size_t)grow * N + n] = val;
        if (Cb) Cb[(size_t)grow * N + n] = f2bf(val);
      }
  }
}

// ---------------- RoPE on q (in place, bf16 [L][H*D]) ----------------
__global__ void rope_q_kernel(unsigned short* __restrict__ q,
                              const float* __restrict__ cosb,
                              const float* __restrict__ sinb) {
  const int total = L_ * H_ * (ROT_ / 2);
  for (int i = blockIdx.x * blockDim.x + threadIdx.x; i < total;
       i += gridDim.x * blockDim.x) {
    const int l   = i / (H_ * (ROT_ / 2));
    const int rem = i - l * (H_ * (ROT_ / 2));
    const int h   = rem / (ROT_ / 2);
    const int p2  = rem - h * (ROT_ / 2);
    const size_t base = (size_t)l * HID_ + h * D_ + 2 * p2;
    const float t1 = bf2f(q[base]), t2 = bf2f(q[base + 1]);
    const float c = cosb[l * (ROT_ / 2) + p2];
    const float s = sinb[l * (ROT_ / 2) + p2];
    q[base]     = f2bf(t1 * c - t2 * s);
    q[base + 1] = f2bf(t2 * c + t1 * s);
  }
}

// ---------------- build K cache: f32 out + bf16 [kvh][key][d] ----------------
__global__ void build_k_kernel(const float* __restrict__ cache_k,
                               const unsigned short* __restrict__ knew,
                               const float* __restrict__ cosb,
                               const float* __restrict__ sinb,
                               float* __restrict__ ck_out,
                               unsigned short* __restrict__ K_bf) {
  const int total = KVH_ * KTOT_ * (D_ / 2);
  for (int i = blockIdx.x * blockDim.x + threadIdx.x; i < total;
       i += gridDim.x * blockDim.x) {
    const int kvh = i / (KTOT_ * (D_ / 2));
    const int rem = i - kvh * (KTOT_ * (D_ / 2));
    const int key = rem / (D_ / 2);
    const int d2  = rem - key * (D_ / 2);
    const int d = 2 * d2;
    float v0, v1;
    if (key < P_) {
      const size_t src = (size_t)kvh * P_ * D_ + (size_t)key * D_ + d;
      v0 = cache_k[src]; v1 = cache_k[src + 1];
    } else {
      const int l = key - P_;
      const size_t src = (size_t)l * (KVH_ * D_) + kvh * D_ + d;
      const float t1 = bf2f(knew[src]), t2 = bf2f(knew[src + 1]);
      if (d2 < ROT_ / 2) {
        const float c = cosb[l * (ROT_ / 2) + d2];
        const float s = sinb[l * (ROT_ / 2) + d2];
        v0 = t1 * c - t2 * s; v1 = t2 * c + t1 * s;
      } else { v0 = t1; v1 = t2; }
    }
    const size_t dst = (size_t)kvh * KTOT_ * D_ + (size_t)key * D_ + d;
    ck_out[dst] = v0; ck_out[dst + 1] = v1;
    K_bf[dst] = f2bf(v0); K_bf[dst + 1] = f2bf(v1);
  }
}

// ---------------- build V cache: f32 out + bf16 transposed [kvh][d][key] ----
__global__ void build_v_kernel(const float* __restrict__ cache_v,
                               const unsigned short* __restrict__ vnew,
                               float* __restrict__ cv_out,
                               unsigned short* __restrict__ Vt_bf) {
  const int total = KVH_ * KTOT_ * D_;
  for (int i = blockIdx.x * blockDim.x + threadIdx.x; i < total;
       i += gridDim.x * blockDim.x) {
    const int kvh = i / (KTOT_ * D_);
    const int rem = i - kvh * (KTOT_ * D_);
    const int key = rem / D_;
    const int d   = rem - key * D_;
    float val;
    if (key < P_) {
      val = cache_v[(size_t)kvh * P_ * D_ + (size_t)key * D_ + d];
    } else {
      val = bf2f(vnew[(size_t)(key - P_) * (KVH_ * D_) + kvh * D_ + d]);
    }
    cv_out[i] = val;
    Vt_bf[(size_t)kvh * D_ * KTOT_ + (size_t)d * KTOT_ + key] = f2bf(val);
  }
}

// ---------------- flash attention ----------------
// 4 waves / block: 4 consecutive heads (same KV head) x one 16-query tile,
// sharing K/V streams (WGP$/L2 locality).
__global__ __launch_bounds__(128)
void attn_kernel(const unsigned short* __restrict__ qb,   // [L][H*D], roped
                 const unsigned short* __restrict__ Kb,   // [KVH][KTOT][D]
                 const unsigned short* __restrict__ Vt,   // [KVH][D][KTOT]
                 unsigned short* __restrict__ ob) {       // [L][H*D]
  __shared__ uint4 p_lds_all[4][64];   // per-wave 16x32 bf16 tile, 16B aligned
  const int wid  = threadIdx.x >> 5;
  const int lane = threadIdx.x & 31;
  const int half = lane >> 4, mr = lane & 15;
  const int qt = blockIdx.x;
  const int h = blockIdx.y * 4 + wid;          // 16 % 4 == 0: same kvh / block
  const int kvh = h / REP_;
  char* pl = (char*)p_lds_all[wid];

  // q tile, A-operand layout, 4 chunks of K=32 over d
  const unsigned qoff =
      (unsigned)((qt * 16 + mr) * HID_ + h * D_) * 2u + 32u * (unsigned)half;
  v16u qa[4];
#pragma unroll
  for (int c = 0; c < 4; ++c) qa[c] = load_a16((const char*)qb, qoff, c * 64);

  // invariant per-lane offsets for K and V B-operands
  unsigned koff[2];
#pragma unroll
  for (int g = 0; g < 2; ++g)
    koff[g] = (unsigned)((g * 16 + mr) * D_) * 2u + 32u * (unsigned)half;
  unsigned voff[8];
#pragma unroll
  for (int t = 0; t < 8; ++t)
    voff[t] = (unsigned)((t * 16 + mr) * KTOT_) * 2u + 32u * (unsigned)half;
  const unsigned ploff = (unsigned)(mr * 64) + 32u * (unsigned)half;

  v8f oacc[8] = {};
  float mrun[8], lrun[8];
#pragma unroll
  for (int r = 0; r < 8; ++r) { mrun[r] = -3.0e38f; lrun[r] = 0.0f; }

  const char* Kbase = (const char*)(Kb + (size_t)kvh * KTOT_ * D_);
  const char* Vbase = (const char*)(Vt + (size_t)kvh * D_ * KTOT_);
  const int kmax = P_ + qt * 16 + 15;          // last visible key
  const int nblk = (kmax + 32) / 32;           // uniform across block (qt only)

  for (int kb = 0; kb < nblk; ++kb) {
    const int key0 = kb * 32;
    const char* kp = Kbase + (size_t)key0 * D_ * 2;   // uniform per iter
    const char* vp = Vbase + (size_t)key0 * 2;
    // ---- S = q @ K^T (16 x 32), 8 wmma ----
    v8f s[2] = {};
#pragma unroll
    for (int g = 0; g < 2; ++g)
#pragma unroll
      for (int c = 0; c < 4; ++c) {
        const v16u bv = load_b16(kp, koff[g], c * 64);
        s[g] = wmma_bf16(qa[c], bv, s[g]);
      }
    // ---- mask + scale + online softmax (per C-layout row r) ----
#pragma unroll
    for (int r = 0; r < 8; ++r) {
      const int qglob = P_ + qt * 16 + r + 8 * half;
      float s0 = s[0][r] * SCALE_F + ((key0 + mr)      <= qglob ? 0.f : NEG_BIG);
      float s1 = s[1][r] * SCALE_F + ((key0 + 16 + mr) <= qglob ? 0.f : NEG_BIG);
      float rm = fmaxf(s0, s1);
      rm = fmaxf(rm, __shfl_xor(rm, 1, 32));
      rm = fmaxf(rm, __shfl_xor(rm, 2, 32));
      rm = fmaxf(rm, __shfl_xor(rm, 4, 32));
      rm = fmaxf(rm, __shfl_xor(rm, 8, 32));   // max over 16-lane row group
      const float newm = fmaxf(mrun[r], rm);
      const float corr = __expf(mrun[r] - newm);
      mrun[r] = newm;
      const float p0 = __expf(s0 - newm);
      const float p1 = __expf(s1 - newm);
      float rs = p0 + p1;
      rs += __shfl_xor(rs, 1, 32);
      rs += __shfl_xor(rs, 2, 32);
      rs += __shfl_xor(rs, 4, 32);
      rs += __shfl_xor(rs, 8, 32);
      lrun[r] = lrun[r] * corr + rs;
#pragma unroll
      for (int t = 0; t < 8; ++t) oacc[t][r] *= corr;
      s[0][r] = p0; s[1][r] = p1;
    }
    // ---- bounce P through LDS: C-layout -> A-layout, as bf16 ----
    __syncthreads();
    unsigned short* plu = (unsigned short*)pl;
#pragma unroll
    for (int r = 0; r < 8; ++r) {
      plu[(r + 8 * half) * 32 + mr]      = f2bf(s[0][r]);
      plu[(r + 8 * half) * 32 + 16 + mr] = f2bf(s[1][r]);
    }
    __syncthreads();
    const v16u pa = load_a16(pl, ploff, 0);   // two ds_load_b128
    // ---- O += P @ V (16 x 128), 8 wmma ----
#pragma unroll
    for (int t = 0; t < 8; ++t) {
      const v16u bv = load_b16(vp, voff[t], 0);
      oacc[t] = wmma_bf16(pa, bv, oacc[t]);
    }
  }
  // ---- normalize + store bf16 [L][H*D] ----
#pragma unroll
  for (int t = 0; t < 8; ++t)
#pragma unroll
    for (int r = 0; r < 8; ++r) {
      const int row = qt * 16 + r + 8 * half;
      const float val = oacc[t][r] / lrun[r];
      ob[(size_t)row * HID_ + h * D_ + t * 16 + mr] = f2bf(val);
    }
}

// ---------------- launch ----------------
extern "C" void kernel_launch(void* const* d_in, const int* in_sizes, int n_in,
                              void* d_out, int out_size, void* d_ws,
                              size_t ws_size, hipStream_t stream) {
  (void)in_sizes; (void)n_in; (void)out_size; (void)ws_size;
  const float* x       = (const float*)d_in[0];
  // d_in[1] = attention_mask (causal mask recomputed in-kernel)
  const float* cosb    = (const float*)d_in[2];
  const float* sinb    = (const float*)d_in[3];
  const float* cache_k = (const float*)d_in[4];
  const float* cache_v = (const float*)d_in[5];
  const float* Wq      = (const float*)d_in[6];
  const float* bq      = (const float*)d_in[7];
  const float* Wk      = (const float*)d_in[8];
  const float* bk      = (const float*)d_in[9];
  const float* Wv      = (const float*)d_in[10];
  const float* bv      = (const float*)d_in[11];
  const float* Wo      = (const float*)d_in[12];

  float* o_out  = (float*)d_out;                    // [L][HID]
  float* ck_out = o_out + (size_t)L_ * HID_;        // [KVH][KTOT][D]
  float* cv_out = ck_out + (size_t)KVH_ * KTOT_ * D_;

  char* ws = (char*)d_ws;
  size_t off = 0;
  auto alloc = [&](size_t bytes) -> void* {
    void* p = ws + off;
    off += (bytes + 255) & ~(size_t)255;
    return p;
  };
  unsigned short* x_bf  = (unsigned short*)alloc((size_t)L_ * HID_ * 2);
  unsigned short* Wqt   = (unsigned short*)alloc((size_t)HID_ * (H_ * D_) * 2);
  unsigned short* Wkt   = (unsigned short*)alloc((size_t)HID_ * (KVH_ * D_) * 2);
  unsigned short* Wvt   = (unsigned short*)alloc((size_t)HID_ * (KVH_ * D_) * 2);
  unsigned short* Wot   = (unsigned short*)alloc((size_t)(H_ * D_) * HID_ * 2);
  unsigned short* q_bf  = (unsigned short*)alloc((size_t)L_ * H_ * D_ * 2);
  unsigned short* knew  = (unsigned short*)alloc((size_t)L_ * KVH_ * D_ * 2);
  unsigned short* vnew  = (unsigned short*)alloc((size_t)L_ * KVH_ * D_ * 2);
  unsigned short* K_bf  = (unsigned short*)alloc((size_t)KVH_ * KTOT_ * D_ * 2);
  unsigned short* Vt_bf = (unsigned short*)alloc((size_t)KVH_ * D_ * KTOT_ * 2);
  unsigned short* attn  = (unsigned short*)alloc((size_t)L_ * H_ * D_ * 2);

  const dim3 b256(256);
  const dim3 wave(32);
  const dim3 attnblk(128);

  // stage bf16 operands (weights transposed to N-major for WMMA B)
  cvt_bf16_kernel  <<<2048, b256, 0, stream>>>(x,  x_bf, L_ * HID_);
  cvt_bf16_T_kernel<<<8192, b256, 0, stream>>>(Wq, Wqt, HID_, 12); // N=4096
  cvt_bf16_T_kernel<<<1024, b256, 0, stream>>>(Wk, Wkt, HID_, 8);  // N=256
  cvt_bf16_T_kernel<<<1024, b256, 0, stream>>>(Wv, Wvt, HID_, 8);  // N=256
  cvt_bf16_T_kernel<<<8192, b256, 0, stream>>>(Wo, Wot, H_ * D_, 12);

  // QKV projections (WMMA, 32x128 tile per wave)
  gemm_bf16_kernel<<<dim3((H_ * D_) / 128, L_ / 32), wave, 0, stream>>>(
      x_bf, Wqt, bq, q_bf, nullptr, L_, H_ * D_, HID_);
  gemm_bf16_kernel<<<dim3((KVH_ * D_) / 128, L_ / 32), wave, 0, stream>>>(
      x_bf, Wkt, bk, knew, nullptr, L_, KVH_ * D_, HID_);
  gemm_bf16_kernel<<<dim3((KVH_ * D_) / 128, L_ / 32), wave, 0, stream>>>(
      x_bf, Wvt, bv, vnew, nullptr, L_, KVH_ * D_, HID_);

  // RoPE + cache assembly (also writes f32 cache outputs)
  rope_q_kernel<<<4096, b256, 0, stream>>>(q_bf, cosb, sinb);
  build_k_kernel<<<(KVH_ * KTOT_ * (D_ / 2)) / 256, b256, 0, stream>>>(
      cache_k, knew, cosb, sinb, ck_out, K_bf);
  build_v_kernel<<<(KVH_ * KTOT_ * D_) / 256, b256, 0, stream>>>(
      cache_v, vnew, cv_out, Vt_bf);

  // flash attention (WMMA): 4 heads (same kvh) per block share K/V streams
  attn_kernel<<<dim3(L_ / 16, H_ / 4), attnblk, 0, stream>>>(
      q_bf, K_bf, Vt_bf, attn);

  // output projection (WMMA, f32 out)
  gemm_bf16_kernel<<<dim3(HID_ / 128, L_ / 32), wave, 0, stream>>>(
      attn, Wot, nullptr, nullptr, o_out, L_, HID_, H_ * D_);
}
